// Scanning_24343874634150
// MI455X (gfx1250) — matrix operand order repaired and checked by reading, not compiled, not run on hardware
//
#include <hip/hip_runtime.h>
#include <hip/hip_bf16.h>
#include <cstdint>

typedef __attribute__((ext_vector_type(16))) _Float16 v16h;
typedef __attribute__((ext_vector_type(8)))  float    v8f;
typedef __attribute__((ext_vector_type(4)))  int      v4i;

#define DIMC     192
#define D_INNER  384
#define DT_RANK  12
#define NB       8
#define LSEQ     4096
#define NTOK     (NB*LSEQ)   // 32768 tokens

#define AS1 __attribute__((address_space(1)))
#define AS3 __attribute__((address_space(3)))

#if defined(__has_builtin)
#if __has_builtin(__builtin_amdgcn_global_load_async_to_lds_b128)
#define HAS_ASYNC 1
#endif
#endif
#ifndef HAS_ASYNC
#define HAS_ASYNC 0
#endif

// 16-byte global -> LDS copy: async DMA path on CDNA5, sync fallback otherwise.
__device__ __forceinline__ void cp16(void* lds, const void* g) {
#if HAS_ASYNC
  __builtin_amdgcn_global_load_async_to_lds_b128(
      (AS1 v4i*)(uintptr_t)g, (AS3 v4i*)(uint32_t)(uintptr_t)lds, 0, 0);
#else
  *(float4*)lds = *(const float4*)g;
#endif
}

#if HAS_ASYNC
#if __has_builtin(__builtin_amdgcn_s_wait_asynccnt)
#define WAIT_ASYNC(n) __builtin_amdgcn_s_wait_asynccnt(n)
#else
#define WAIT_ASYNC(n) asm volatile("s_wait_asynccnt %0" :: "i"(n) : "memory")
#endif
#else
#define WAIT_ASYNC(n) ((void)0)
#endif

// ---- WMMA fragment helpers (layouts from CDNA5 ISA 7.12.2, 16-bit data) ----
__device__ __forceinline__ v16h load_a_frag(const _Float16* a, int kw, int kc, int lane) {
  int m = lane & 15, half = lane >> 4;
  v16h r;
#pragma unroll
  for (int j = 0; j < 16; ++j) {
    int v = j >> 1;
    int base = (v < 4) ? (2 * v) : (16 + 2 * (v - 4));
    int k = kc * 32 + base + half * 8 + (j & 1);
    r[j] = a[m * kw + k];
  }
  return r;
}

__device__ __forceinline__ v16h load_b_frag(const _Float16* bf, int kc, int nt, int ntiles, int lane) {
  const _Float16* p = bf + (((size_t)(kc * ntiles + nt)) * 32 + lane) * 16;
  v16h r;
#pragma unroll
  for (int j = 0; j < 16; ++j) r[j] = p[j];
  return r;
}

// Pack row-major W[n][k] (fp32) into f16 B fragments, zero-padded to 16x32 tiles.
__global__ void pack_frags(const float* __restrict__ W, int Nreal, int Kreal, int NT,
                           _Float16* __restrict__ dst) {
  int tile = blockIdx.x;          // kc*NT + nt
  int kc = tile / NT, nt = tile % NT;
  int lane = threadIdx.x;
  _Float16* p = dst + ((size_t)tile * 32 + lane) * 16;
  int half = lane >> 4, n = nt * 16 + (lane & 15);
#pragma unroll
  for (int j = 0; j < 16; ++j) {
    int v = j >> 1;
    int k = kc * 32 + ((v < 4) ? (2 * v) : (16 + 2 * (v - 4))) + (j & 1) + half * 8;
    float val = (k < Kreal && n < Nreal) ? W[(size_t)n * Kreal + k] : 0.f;
    p[j] = (_Float16)val;
  }
}

// ---- Kernel 1: dual LayerNorm + input projections (x = LN1@Wx^T, z = LN2@Wz^T) ----
__global__ __launch_bounds__(256)
void ln_in_proj(const float* __restrict__ xe,
                const float* __restrict__ g1, const float* __restrict__ b1,
                const float* __restrict__ g2, const float* __restrict__ b2,
                const _Float16* __restrict__ wxF, const _Float16* __restrict__ wzF,
                float* __restrict__ xout, float* __restrict__ zout) {
  __shared__ __align__(16) float    tokT[DIMC * 16];   // [c][m] (async-copy friendly)
  __shared__ _Float16 a1[16 * DIMC];
  __shared__ _Float16 a2[16 * DIMC];
  int tid = threadIdx.x;
  int g0 = blockIdx.x * 16;
  int b = g0 >> 12, hw0 = g0 & 4095;

  // tokT[c][m] = xe[b][c][hw0+m]; 16B-aligned contiguous segments -> async DMA
  for (int i = tid; i < DIMC * 4; i += 256) {
    int c = i >> 2, seg = i & 3;
    cp16(&tokT[c * 16 + seg * 4], xe + ((size_t)b * DIMC + c) * 4096 + hw0 + seg * 4);
  }
  WAIT_ASYNC(0);
  __syncthreads();

  int wv = tid >> 5, lane = tid & 31;
  {
    int m = 2 * wv + (lane >> 4);   // half-wave per token
    int r = lane & 15;
    float s = 0.f, ss = 0.f;
#pragma unroll
    for (int q = 0; q < 12; ++q) { float v = tokT[(r + 16 * q) * 16 + m]; s += v; ss += v * v; }
#pragma unroll
    for (int off = 8; off; off >>= 1) { s += __shfl_xor(s, off, 16); ss += __shfl_xor(ss, off, 16); }
    float mu  = s  * (1.f / 192.f);
    float var = ss * (1.f / 192.f) - mu * mu;
    float rstd = rsqrtf(var + 1e-5f);
#pragma unroll
    for (int q = 0; q < 12; ++q) {
      int c = r + 16 * q;
      float xn = (tokT[c * 16 + m] - mu) * rstd;
      a1[m * DIMC + c] = (_Float16)(xn * g1[c] + b1[c]);
      a2[m * DIMC + c] = (_Float16)(xn * g2[c] + b2[c]);
    }
  }
  __syncthreads();

  // 48 output tiles (24 x-tiles + 24 z-tiles) over 8 waves
  for (int i = 0; i < 6; ++i) {
    int ti = wv * 6 + i;
    bool isx = ti < 24;
    int nt = isx ? ti : ti - 24;
    const _Float16* afr = isx ? a1 : a2;
    const _Float16* bfr = isx ? wxF : wzF;
    float* dst = isx ? xout : zout;
    v8f acc = {};
#pragma unroll
    for (int kc = 0; kc < 6; ++kc) {
      v16h a  = load_a_frag(afr, DIMC, kc, lane);
      v16h bb = load_b_frag(bfr, kc, nt, 24, lane);
      acc = __builtin_amdgcn_wmma_f32_16x16x32_f16(false, a, false, bb, (short)0, acc, false, false);
    }
    int n = lane & 15, half = lane >> 4;
#pragma unroll
    for (int r = 0; r < 8; ++r) {
      int m = r + 8 * half;
      dst[((size_t)(g0 + m)) * D_INNER + nt * 16 + n] = acc[r];
    }
  }
}

// ---- Kernel 2: per-branch causal conv + SiLU, x-proj GEMM (->44), dt GEMM + softplus ----
__global__ __launch_bounds__(256)
void conv_xproj(const float* __restrict__ x,
                const float* __restrict__ cwF, const float* __restrict__ cbF,
                const float* __restrict__ cwB, const float* __restrict__ cbB,
                const _Float16* __restrict__ wxp0, const _Float16* __restrict__ wxp1,
                const _Float16* __restrict__ wdt0, const _Float16* __restrict__ wdt1,
                const float* __restrict__ dtb0, const float* __restrict__ dtb1,
                float* __restrict__ xc, float* __restrict__ dt,
                float* __restrict__ bc, float* __restrict__ ccv) {
  __shared__ _Float16 afr[16 * D_INNER];
  __shared__ float    dbl[16 * 48];
  __shared__ _Float16 a2[16 * 32];
  int tid = threadIdx.x;
  int br = blockIdx.y;                    // 0 = forward, 1 = backward
  int g0 = blockIdx.x * 16;
  int b = g0 >> 12, pb0 = g0 & 4095;
  const float* cw = br ? cwB : cwF;
  const float* cb = br ? cbB : cbF;
  size_t brOff = (size_t)br * NTOK;

  // depthwise causal conv (in branch direction) + SiLU
  for (int i = tid; i < 16 * D_INNER; i += 256) {
    int m = i / D_INNER, d = i % D_INNER;
    int pb = pb0 + m;
    float s = cb[d];
#pragma unroll
    for (int kk = 0; kk < 4; ++kk) {
      int q = pb - 3 + kk;
      if (q >= 0) {
        int orig = br ? (4095 - q) : q;
        s += x[((size_t)b * 4096 + orig) * D_INNER + d] * cw[d * 4 + kk];
      }
    }
    s = s / (1.f + __expf(-s));
    xc[(brOff + g0 + m) * D_INNER + d] = s;
    afr[m * D_INNER + d] = (_Float16)s;
  }
  __syncthreads();

  int wv = tid >> 5, lane = tid & 31;
  const _Float16* wxp = br ? wxp1 : wxp0;
  if (wv < 3) {                           // dbl = xc @ Wxproj^T  (N padded 44->48)
    int nt = wv;
    v8f acc = {};
#pragma unroll
    for (int kc = 0; kc < 12; ++kc) {
      v16h a  = load_a_frag(afr, D_INNER, kc, lane);
      v16h bb = load_b_frag(wxp, kc, nt, 3, lane);
      acc = __builtin_amdgcn_wmma_f32_16x16x32_f16(false, a, false, bb, (short)0, acc, false, false);
    }
    int n = lane & 15, half = lane >> 4;
#pragma unroll
    for (int r = 0; r < 8; ++r) dbl[(r + 8 * half) * 48 + nt * 16 + n] = acc[r];
  }
  __syncthreads();

  // split dbl -> dt-input (K padded 12->32), Bc, Cc
  for (int i = tid; i < 16 * 32; i += 256) {
    int m = i >> 5, k = i & 31;
    a2[i] = (_Float16)((k < DT_RANK) ? dbl[m * 48 + k] : 0.f);
    float v = dbl[m * 48 + 12 + k];
    size_t idx = (brOff + g0 + m) * 16;
    if (k < 16) bc[idx + k] = v;
    else        ccv[idx + (k - 16)] = v;
  }
  __syncthreads();

  const _Float16* wdt = br ? wdt1 : wdt0;
  const float*    dtb = br ? dtb1 : dtb0;
  for (int nt = wv; nt < 24; nt += 8) {   // dt = softplus(dbl[:, :12] @ Wdt^T + bias)
    v8f acc = {};
    v16h a  = load_a_frag(a2, 32, 0, lane);
    v16h bb = load_b_frag(wdt, 0, nt, 24, lane);
    acc = __builtin_amdgcn_wmma_f32_16x16x32_f16(false, a, false, bb, (short)0, acc, false, false);
    int n = lane & 15, half = lane >> 4;
#pragma unroll
    for (int r = 0; r < 8; ++r) {
      int m = r + 8 * half;
      int col = nt * 16 + n;
      float val = acc[r] + dtb[col];
      float sp = (val > 20.f) ? val : __logf(1.f + __expf(val));
      dt[(brOff + g0 + m) * D_INNER + col] = sp;
    }
  }
}

// ---- Kernel 3: selective scan with double-buffered async LDS pipeline ----
// One lane per (branch,batch,channel); 16 states in VGPRs. 128 chunks of T=32 steps.
// Per chunk/wave: 18 async b128 loads -> s_wait_asynccnt 18 overlaps next chunk with compute.
__device__ __forceinline__ void scan_issue(const float* dtg, const float* xcg,
                                           const float* bcg, const float* ccg,
                                           float* sdt, float* sxc, float* sbc, float* scc,
                                           int ck, int buf, int tid) {
  const size_t row0 = (size_t)ck * 32;
  float* d0 = sdt + buf * 4096;
  float* x0 = sxc + buf * 4096;
  for (int i = tid; i < 1024; i += 128) {
    int t = i >> 5, q = (i & 31) * 4;
    size_t go = (row0 + t) * D_INNER + q;
    cp16(&d0[t * 128 + q], dtg + go);
    cp16(&x0[t * 128 + q], xcg + go);
  }
  cp16(sbc + buf * 512 + tid * 4, bcg + row0 * 16 + tid * 4);
  cp16(scc + buf * 512 + tid * 4, ccg + row0 * 16 + tid * 4);
}

__global__ __launch_bounds__(128)
void scan_kernel(const float* __restrict__ xc, const float* __restrict__ dt,
                 const float* __restrict__ bc, const float* __restrict__ ccv,
                 const float* __restrict__ AlF, const float* __restrict__ AlB,
                 const float* __restrict__ DFp, const float* __restrict__ DBp,
                 float* __restrict__ y) {
  extern __shared__ __align__(16) float smem[];
  float* sdt = smem;                  // 2 * 32*128
  float* sxc = smem + 8192;           // 2 * 32*128
  float* sbc = smem + 16384;          // 2 * 32*16
  float* scc = smem + 17408;          // 2 * 32*16
  int tid = threadIdx.x;
  int br  = blockIdx.x / (NB * 3);
  int rem = blockIdx.x % (NB * 3);
  int b = rem / 3, cg = rem % 3;
  int d = cg * 128 + tid;

  const float* Alog = br ? AlB : AlF;
  float A[16], h[16];
#pragma unroll
  for (int n = 0; n < 16; ++n) { A[n] = -__expf(Alog[d * 16 + n]); h[n] = 0.f; }
  float Dd = (br ? DBp : DFp)[d];

  size_t rowBase = (size_t)br * NTOK + (size_t)b * 4096;
  const float* dtg = dt  + rowBase * D_INNER + cg * 128;
  const float* xcg = xc  + rowBase * D_INNER + cg * 128;
  const float* bcg = bc  + rowBase * 16;
  const float* ccg = ccv + rowBase * 16;
  float* yg = y + rowBase * D_INNER + d;

  int buf = 0;
  scan_issue(dtg, xcg, bcg, ccg, sdt, sxc, sbc, scc, 0, 0, tid);
  for (int ck = 0; ck < 128; ++ck) {
    bool more = (ck + 1) < 128;
    if (more) scan_issue(dtg, xcg, bcg, ccg, sdt, sxc, sbc, scc, ck + 1, buf ^ 1, tid);
    if (more) { WAIT_ASYNC(18); } else { WAIT_ASYNC(0); }
    __syncthreads();                       // chunk ck resident for all waves
    const float* d0 = sdt + buf * 4096;
    const float* x0 = sxc + buf * 4096;
    const float* b0 = sbc + buf * 512;
    const float* c0 = scc + buf * 512;
    for (int t = 0; t < 32; ++t) {
      int pb = ck * 32 + t;
      float dtv = d0[t * 128 + tid];
      float xv  = x0[t * 128 + tid];
      float dx = dtv * xv;
      float yv = 0.f;
#pragma unroll
      for (int n = 0; n < 16; ++n) {
        float dA = __expf(dtv * A[n]);
        h[n] = fmaf(h[n], dA, dx * b0[t * 16 + n]);
        yv = fmaf(h[n], c0[t * 16 + n], yv);
      }
      int orig = br ? (4095 - pb) : pb;    // un-reverse backward branch on store
      yg[(size_t)orig * D_INNER] = yv + xv * Dd;
    }
    __syncthreads();                       // done reading buf before it is refilled
    buf ^= 1;
  }
}

// ---- Kernel 4: gate with SiLU(z), output GEMM (384->192) + bias, transposed store ----
__global__ __launch_bounds__(256)
void out_proj(const float* __restrict__ y, const float* __restrict__ z,
              const _Float16* __restrict__ woF, const float* __restrict__ bout,
              float* __restrict__ out) {
  __shared__ _Float16 afr[16 * D_INNER];
  int tid = threadIdx.x;
  int g0 = blockIdx.x * 16;
  int b = g0 >> 12, hw0 = g0 & 4095;

  for (int i = tid; i < 16 * D_INNER; i += 256) {
    int m = i / D_INNER, d = i % D_INNER;
    size_t base = ((size_t)g0 + m) * D_INNER + d;
    float yv = y[base] + y[(size_t)NTOK * D_INNER + base];   // y_f + y_b
    float zv = z[base];
    float sz = zv / (1.f + __expf(-zv));
    afr[m * D_INNER + d] = (_Float16)(yv * sz);
  }
  __syncthreads();

  int wv = tid >> 5, lane = tid & 31;
  for (int nt = wv; nt < 12; nt += 8) {
    v8f acc = {};
#pragma unroll
    for (int kc = 0; kc < 12; ++kc) {
      v16h a  = load_a_frag(afr, D_INNER, kc, lane);
      v16h bb = load_b_frag(woF, kc, nt, 12, lane);
      acc = __builtin_amdgcn_wmma_f32_16x16x32_f16(false, a, false, bb, (short)0, acc, false, false);
    }
    int n = lane & 15, half = lane >> 4;
    int col = nt * 16 + n;
    float bo = bout[col];
#pragma unroll
    for (int r = 0; r < 8; ++r) {
      int m = r + 8 * half;
      out[((size_t)b * DIMC + col) * 4096 + hw0 + m] = acc[r] + bo;
    }
  }
}

extern "C" void kernel_launch(void* const* d_in, const int* in_sizes, int n_in,
                              void* d_out, int out_size, void* d_ws, size_t ws_size,
                              hipStream_t stream) {
  (void)in_sizes; (void)n_in; (void)out_size; (void)ws_size;
  const float* xe   = (const float*)d_in[0];
  const float* g1   = (const float*)d_in[1];
  const float* b1   = (const float*)d_in[2];
  const float* g2   = (const float*)d_in[3];
  const float* b2   = (const float*)d_in[4];
  const float* Winx = (const float*)d_in[5];
  const float* Winz = (const float*)d_in[6];
  const float* Wout = (const float*)d_in[7];
  const float* bout = (const float*)d_in[8];
  const float* cwF  = (const float*)d_in[9];
  const float* cbF  = (const float*)d_in[10];
  const float* WxpF = (const float*)d_in[11];
  const float* WdtF = (const float*)d_in[12];
  const float* dtbF = (const float*)d_in[13];
  const float* AlF  = (const float*)d_in[14];
  const float* DFp  = (const float*)d_in[15];
  const float* cwB  = (const float*)d_in[16];
  const float* cbB  = (const float*)d_in[17];
  const float* WxpB = (const float*)d_in[18];
  const float* WdtB = (const float*)d_in[19];
  const float* dtbB = (const float*)d_in[20];
  const float* AlB  = (const float*)d_in[21];
  const float* DBp  = (const float*)d_in[22];

  size_t NT384 = (size_t)NTOK * D_INNER;
  float* ws   = (float*)d_ws;
  float* xbuf = ws;                        ws += NT384;
  float* zbuf = ws;                        ws += NT384;
  float* xcb  = ws;                        ws += 2 * NT384;
  float* dtbf = ws;                        ws += 2 * NT384;
  float* bcb  = ws;                        ws += 2 * (size_t)NTOK * 16;
  float* ccb  = ws;                        ws += 2 * (size_t)NTOK * 16;
  float* ybuf = ws;                        ws += 2 * NT384;
  _Float16* fr = (_Float16*)ws;
  _Float16* wxFr   = fr; fr += 6 * 24 * 512;
  _Float16* wzFr   = fr; fr += 6 * 24 * 512;
  _Float16* wxpFr0 = fr; fr += 12 * 3 * 512;
  _Float16* wxpFr1 = fr; fr += 12 * 3 * 512;
  _Float16* wdtFr0 = fr; fr += 1 * 24 * 512;
  _Float16* wdtFr1 = fr; fr += 1 * 24 * 512;
  _Float16* woFr   = fr; fr += 12 * 12 * 512;

  // Pre-pack weights into WMMA B-fragment order (f16, zero-padded tiles)
  pack_frags<<<6 * 24, 32, 0, stream>>>(Winx, 384, 192, 24, wxFr);
  pack_frags<<<6 * 24, 32, 0, stream>>>(Winz, 384, 192, 24, wzFr);
  pack_frags<<<12 * 3, 32, 0, stream>>>(WxpF, 44, 384, 3, wxpFr0);
  pack_frags<<<12 * 3, 32, 0, stream>>>(WxpB, 44, 384, 3, wxpFr1);
  pack_frags<<<1 * 24, 32, 0, stream>>>(WdtF, 384, 12, 24, wdtFr0);
  pack_frags<<<1 * 24, 32, 0, stream>>>(WdtB, 384, 12, 24, wdtFr1);
  pack_frags<<<12 * 12, 32, 0, stream>>>(Wout, 192, 384, 12, woFr);

  ln_in_proj<<<NTOK / 16, 256, 0, stream>>>(xe, g1, b1, g2, b2, wxFr, wzFr, xbuf, zbuf);
  conv_xproj<<<dim3(NTOK / 16, 2), 256, 0, stream>>>(xbuf, cwF, cbF, cwB, cbB,
      wxpFr0, wxpFr1, wdtFr0, wdtFr1, dtbF, dtbB, xcb, dtbf, bcb, ccb);
  scan_kernel<<<2 * NB * 3, 128, 72 * 1024, stream>>>(xcb, dtbf, bcb, ccb,
      AlF, AlB, DFp, DBp, ybuf);
  out_proj<<<NTOK / 16, 256, 0, stream>>>(ybuf, zbuf, woFr, bout, (float*)d_out);
}
